// RelationGraphAttention_472446402571
// MI455X (gfx1250) — compile-verified, matrix operand order checked by reading, not run on hardware
//
#include <hip/hip_runtime.h>

typedef unsigned short u16;
typedef unsigned int   u32;
typedef unsigned long long u64;

typedef __attribute__((ext_vector_type(16))) __bf16 bf16x16;
typedef __attribute__((ext_vector_type(8)))  __bf16 bf16x8;
typedef __attribute__((ext_vector_type(8)))  float  f32x8;
typedef __attribute__((ext_vector_type(4)))  u32    v4u;
typedef __attribute__((ext_vector_type(8)))  int    v8i;
typedef __attribute__((ext_vector_type(4)))  int    v4i;

#define HID   768
#define SEQ   1024
#define TLEN  512
#define BATCH 8
#define HEADS 12
#define HD    64
#define MTOT  (BATCH * SEQ)          // 8192 rows
#define WELEM (HID * HID)            // 589824 per weight matrix

// ---- workspace layout (bytes) ----
#define OFF_REL  ((size_t)0)
#define OFF_X    ((size_t)64)
#define SZ_X     ((size_t)MTOT * HID * 2)          // 12,582,912
#define OFF_W    (OFF_X + SZ_X)
#define SZ_W     ((size_t)4 * WELEM * 2)           // 4,718,592
#define OFF_Q    (OFF_W + SZ_W)
#define OFF_K    (OFF_Q + SZ_X)
#define OFF_VT   (OFF_K + SZ_X)
#define OFF_CTX  (OFF_VT + SZ_X)

union FragA { bf16x16 v; bf16x8 h[2]; };

__device__ __forceinline__ u16 f2bf(float f) {
  u32 u = __builtin_bit_cast(u32, f);
  u += 0x7FFFu + ((u >> 16) & 1u);              // round-to-nearest-even
  return (u16)(u >> 16);
}

__device__ __forceinline__ f32x8 wmma_bf16(bf16x16 a, bf16x16 b, f32x8 c) {
  return __builtin_amdgcn_wmma_f32_16x16x32_bf16(false, a, false, b, (short)0, c,
                                                 false, false);
}

// Load a 16x32 bf16 fragment (A layout; B layout identical with lane = N) from a
// row-major matrix (global or LDS; force-inlined so addrspace is inferred).
// ISA layout: lane<16 holds K {0..7, 16..23}, lane>=16 holds K {8..15, 24..31}.
__device__ __forceinline__ bf16x16 load_frag(const u16* p, int ld, int r0, int c0,
                                             int lane) {
  int r = r0 + (lane & 15);
  int c = c0 + ((lane >> 4) << 3);
  const u16* base = p + (size_t)r * ld + c;
  FragA f;
  f.h[0] = *(const bf16x8*)(base);
  f.h[1] = *(const bf16x8*)(base + 16);
  return f.v;
}

// Async global -> LDS copy of 16 bytes (CDNA5 GLOBAL_LOAD_ASYNC_TO_LDS_B128).
__device__ __forceinline__ void async_b128(u32 lds_off, const u16* gptr) {
  u64 ga = (u64)(uintptr_t)gptr;
  asm volatile("global_load_async_to_lds_b128 %0, %1, off"
               :: "v"(lds_off), "v"(ga) : "memory");
}
__device__ __forceinline__ void wait_async_0() {
  asm volatile("s_wait_asynccnt 0x0" ::: "memory");
}
__device__ __forceinline__ void wait_async_3() {
  asm volatile("s_wait_asynccnt 0x3" ::: "memory");
}

#if __has_builtin(__builtin_amdgcn_tensor_load_to_lds)
#define HAVE_TDM 1
// One-wave DMA of a 2-D bf16 tile (tile_d0 elems per row, tile_d1 rows,
// row stride in elements) from global into LDS at lds_off (packed row-major).
// D# bitfields per CDNA5 ISA; groups 2/3 unused.
__device__ __forceinline__ void tdm_load_2d(u32 lds_off, const u16* gptr,
                                            int tile_d0, int tile_d1,
                                            int stride_elems) {
  u64 ga = (u64)(uintptr_t)gptr;
  v4u g0;
  g0[0] = 1u;                                     // count=1, user descriptor
  g0[1] = lds_off;                                // lds_addr (bytes)
  g0[2] = (u32)(ga & 0xffffffffu);                // global_addr[31:0]
  g0[3] = (u32)((ga >> 32) & 0x1ffffffu) | (2u << 30);  // addr[56:32] | type=2
  v8i g1;
  g1[0] = (1 << 16);                              // data_size = 1 -> 2 bytes
  g1[1] = (tile_d0 & 0xffff) << 16;               // tensor_dim0[15:0]
  g1[2] = ((u32)tile_d0 >> 16) | ((tile_d1 & 0xffff) << 16);  // dim0 hi | dim1 lo
  g1[3] = ((u32)tile_d1 >> 16) | ((tile_d0 & 0xffff) << 16);  // dim1 hi | tile_dim0
  g1[4] = (tile_d1 & 0xffff);                     // tile_dim1 (tile_dim2 = 0)
  g1[5] = stride_elems;                           // tensor_dim0_stride[31:0]
  g1[6] = 0;                                      // stride hi | dim1_stride lo
  g1[7] = 0;
  v4i z4 = {0, 0, 0, 0};
#if __clang_major__ >= 23
  v8i z8 = {0, 0, 0, 0, 0, 0, 0, 0};
  __builtin_amdgcn_tensor_load_to_lds(g0, g1, z4, z4, z8, 0);
#else
  __builtin_amdgcn_tensor_load_to_lds(g0, g1, z4, z4, 0);
#endif
}
#else
#define HAVE_TDM 0
#endif

// ---------------------------------------------------------------------------
// relation-type bias: mean over hidden dim of each rel_emb row -> 3 floats
__global__ void rga_relmean_kernel(const float* __restrict__ rel_emb,
                                   float* __restrict__ relmean) {
  __shared__ float red[256];
  int t = threadIdx.x;
  const float* row = rel_emb + blockIdx.x * HID;
  float s = 0.f;
  for (int i = t; i < HID; i += 256) s += row[i];
  red[t] = s;
  __syncthreads();
  for (int off = 128; off > 0; off >>= 1) {
    if (t < off) red[t] += red[t + off];
    __syncthreads();
  }
  if (t == 0) relmean[blockIdx.x] = red[0] / (float)HID;
}

// concat(text,image) fp32 -> bf16, [B,S,HID] row-major
__global__ void rga_convert_x_kernel(const float* __restrict__ text,
                                     const float* __restrict__ image,
                                     u16* __restrict__ xbf) {
  size_t idx = ((size_t)blockIdx.x * 256 + threadIdx.x) * 4;
  if (idx >= (size_t)MTOT * HID) return;
  int m = (int)(idx / HID), c = (int)(idx % HID);
  int b = m >> 10, s = m & 1023;
  const float* src = (s < TLEN)
      ? (text  + ((size_t)(b * TLEN + s) * HID) + c)
      : (image + ((size_t)(b * TLEN + (s - TLEN)) * HID) + c);
  float4 v = *(const float4*)src;
  u16* dst = xbf + idx;
  dst[0] = f2bf(v.x); dst[1] = f2bf(v.y); dst[2] = f2bf(v.z); dst[3] = f2bf(v.w);
}

// Wq|Wk|Wv|Wo fp32 -> bf16 concatenated
__global__ void rga_convert_w_kernel(const float* __restrict__ Wq,
                                     const float* __restrict__ Wk,
                                     const float* __restrict__ Wv,
                                     const float* __restrict__ Wo,
                                     u16* __restrict__ wbf) {
  size_t idx = ((size_t)blockIdx.x * 256 + threadIdx.x) * 4;
  if (idx >= (size_t)4 * WELEM) return;
  int which = (int)(idx / WELEM);
  size_t r = idx % WELEM;
  const float* src = (which == 0 ? Wq : which == 1 ? Wk : which == 2 ? Wv : Wo) + r;
  float4 v = *(const float4*)src;
  u16* dst = wbf + idx;
  dst[0] = f2bf(v.x); dst[1] = f2bf(v.y); dst[2] = f2bf(v.z); dst[3] = f2bf(v.w);
}

// ---------------------------------------------------------------------------
// y = A[8192,768](bf16) * W[768,768]^T(bf16) + bias  (torch Linear convention)
// MODE 0: store bf16 row-major [M,N]
// MODE 1: store bf16 transposed per head -> [B, H, D, S]  (for V)
// MODE 2: store fp32 row-major [M,N]                      (final output)
// Block: 256 threads = 8 waves; block tile 128(M) x 64(N); wave tile 32x32.
// Double-buffered staging via TDM (one wave DMAs both tiles) or async-b128.
template <int MODE>
__global__ void rga_gemm_kernel(const u16* __restrict__ A, const u16* __restrict__ W,
                                const float* __restrict__ bias,
                                void* __restrict__ outp) {
  constexpr int K = HID, N = HID;
  constexpr int KSTEPS = K / 32;                  // 24
  __shared__ __align__(16) u16 As[2][128 * 32];   // 2 x 8 KB
  __shared__ __align__(16) u16 Bs[2][64 * 32];    // 2 x 4 KB
  int tid = threadIdx.x;
  int lane = tid & 31;
  int wave = tid >> 5;
  int wr = wave >> 1, wc = wave & 1;
  int m0 = blockIdx.x * 128;
  int n0 = blockIdx.y * 64;

  u32 asOff = (u32)(uintptr_t)&As[0][0];
  u32 bsOff = (u32)(uintptr_t)&Bs[0][0];

  auto issue_tile = [&](int buf, int k0) {
#if HAVE_TDM
    if (wave == 0) {        // TDM issues once per wave, EXEC-independent
      tdm_load_2d(asOff + buf * 8192, A + (size_t)m0 * K + k0, 32, 128, K);
      tdm_load_2d(bsOff + buf * 4096, W + (size_t)n0 * K + k0, 32, 64, K);
    }
#else
#pragma unroll
    for (int i = 0; i < 2; i++) {
      int id = tid + i * 256;
      int row = id >> 2, ch = id & 3;
      async_b128(asOff + buf * 8192 + id * 16,
                 A + (size_t)(m0 + row) * K + k0 + ch * 8);
    }
    {
      int row = tid >> 2, ch = tid & 3;
      async_b128(bsOff + buf * 4096 + tid * 16,
                 W + (size_t)(n0 + row) * K + k0 + ch * 8);
    }
#endif
  };
  auto wait_prev = [&](bool last) {
#if HAVE_TDM
    if (wave == 0) {
      if (last) __builtin_amdgcn_s_wait_tensorcnt((short)0);
      else      __builtin_amdgcn_s_wait_tensorcnt((short)2);
    }
#else
    if (last) wait_async_0(); else wait_async_3();
#endif
  };

  f32x8 acc[2][2] = {};
  auto compute = [&](int cur) {
    const u16* Ab = &As[cur][0];
    const u16* Bb = &Bs[cur][0];
    bf16x16 a0 = load_frag(Ab, 32, wr * 32,      0, lane);
    bf16x16 a1 = load_frag(Ab, 32, wr * 32 + 16, 0, lane);
    bf16x16 b0 = load_frag(Bb, 32, wc * 32,      0, lane);
    bf16x16 b1 = load_frag(Bb, 32, wc * 32 + 16, 0, lane);
    acc[0][0] = wmma_bf16(a0, b0, acc[0][0]);
    acc[0][1] = wmma_bf16(a0, b1, acc[0][1]);
    acc[1][0] = wmma_bf16(a1, b0, acc[1][0]);
    acc[1][1] = wmma_bf16(a1, b1, acc[1][1]);
  };

  issue_tile(0, 0);
#pragma unroll 2
  for (int it = 0; it < KSTEPS - 1; ++it) {       // branch-free steady state
    issue_tile((it + 1) & 1, (it + 1) * 32);
    wait_prev(false);
    __syncthreads();                              // tile visible to all waves
    compute(it & 1);
    __syncthreads();                              // reads done before overwrite
  }
  wait_prev(true);                                // epilogue
  __syncthreads();
  compute((KSTEPS - 1) & 1);

  int lrow = 8 * (lane >> 4);
  int lcol = lane & 15;
#pragma unroll
  for (int i = 0; i < 2; i++)
#pragma unroll
    for (int j = 0; j < 2; j++) {
      int col = n0 + wc * 32 + j * 16 + lcol;
      float bv = bias[col];
#pragma unroll
      for (int g = 0; g < 8; g++) {
        int row = m0 + wr * 32 + i * 16 + g + lrow;
        float v = acc[i][j][g] + bv;
        if (MODE == 0) {
          ((u16*)outp)[(size_t)row * N + col] = f2bf(v);
        } else if (MODE == 1) {
          int b = row >> 10, s = row & 1023;       // row = b*SEQ+s, col = h*64+d
          ((u16*)outp)[(((size_t)b * HID + col) << 10) + s] = f2bf(v);
        } else {
          ((float*)outp)[(size_t)row * N + col] = v;
        }
      }
    }
}

// ---------------------------------------------------------------------------
// Flash attention with per-tile relation bias.
// grid: (S/128, B*HEADS). 8 waves/block, each wave owns 16 query rows.
// K/V tiles for each 32-key step staged once per block via async LDS copies.
__global__ void rga_attn_kernel(const u16* __restrict__ Q, const u16* __restrict__ Kb,
                                const u16* __restrict__ Vt,
                                const float* __restrict__ relmean,
                                u16* __restrict__ ctx) {
  __shared__ __align__(16) u16 Plds[8][16][32];   // per-wave P staging, 8 KB
  __shared__ __align__(16) u16 Ks[32 * 64];       // 32 keys x 64 dims, 4 KB
  __shared__ __align__(16) u16 Vs[64 * 32];       // 64 dims x 32 keys, 4 KB
  int tid = threadIdx.x;
  int lane = tid & 31;
  int w = tid >> 5;
  int bh = blockIdx.y;
  int b = bh / HEADS, h = bh % HEADS;
  int q0 = blockIdx.x * 128 + w * 16;

  u32 ksOff = (u32)(uintptr_t)&Ks[0];
  u32 vsOff = (u32)(uintptr_t)&Vs[0];

  float r0v = relmean[0], r1v = relmean[1], r2v = relmean[2];
  int qside = (q0 >= TLEN);

  bf16x16 aq0 = load_frag(Q, HID, b * SEQ + q0, h * HD, lane);
  bf16x16 aq1 = load_frag(Q, HID, b * SEQ + q0, h * HD + 32, lane);
  const u16* Kh = Kb + (size_t)(b * SEQ) * HID + h * HD;      // rows: keys
  const u16* Vth = Vt + (size_t)(b * HEADS + h) * HD * SEQ;   // [D=64][S=1024]

  f32x8 O[4] = {};
  float mrun[8], lrun[8];
#pragma unroll
  for (int g = 0; g < 8; g++) { mrun[g] = -1e30f; lrun[g] = 0.f; }

  for (int kk = 0; kk < SEQ; kk += 32) {
    // stage K tile: 32 rows x 128 B  (thread t: row=t>>3, chunk=t&7)
    {
      int row = tid >> 3, ch = tid & 7;
      async_b128(ksOff + tid * 16, Kh + (size_t)(kk + row) * HID + ch * 8);
    }
    // stage V tile: 64 rows x 64 B   (thread t: row=t>>2, chunk=t&3)
    {
      int row = tid >> 2, ch = tid & 3;
      async_b128(vsOff + tid * 16, Vth + (size_t)row * SEQ + kk + ch * 8);
    }
    wait_async_0();
    __syncthreads();

    f32x8 s0 = {}, s1 = {};
    s0 = wmma_bf16(aq0, load_frag(Ks, 64, 0,  0,  lane), s0);
    s0 = wmma_bf16(aq1, load_frag(Ks, 64, 0,  32, lane), s0);
    s1 = wmma_bf16(aq0, load_frag(Ks, 64, 16, 0,  lane), s1);
    s1 = wmma_bf16(aq1, load_frag(Ks, 64, 16, 32, lane), s1);

    int ks0 = (kk >= TLEN), ks1 = (kk + 16 >= TLEN);
    float bias0 = qside ? (ks0 ? r0v : r2v) : (ks0 ? r1v : r0v);
    float bias1 = qside ? (ks1 ? r0v : r2v) : (ks1 ? r1v : r0v);

#pragma unroll
    for (int g = 0; g < 8; g++) {
      float v0 = s0[g] * 0.125f + bias0;
      float v1 = s1[g] * 0.125f + bias1;
      float rm = fmaxf(v0, v1);
#pragma unroll
      for (int msk = 1; msk < 16; msk <<= 1) rm = fmaxf(rm, __shfl_xor(rm, msk, 32));
      float mnew = fmaxf(mrun[g], rm);
      float alpha = __expf(mrun[g] - mnew);
      float p0 = __expf(v0 - mnew);
      float p1 = __expf(v1 - mnew);
      float rs = p0 + p1;
#pragma unroll
      for (int msk = 1; msk < 16; msk <<= 1) rs += __shfl_xor(rs, msk, 32);
      lrun[g] = lrun[g] * alpha + rs;
      mrun[g] = mnew;
      O[0][g] *= alpha; O[1][g] *= alpha; O[2][g] *= alpha; O[3][g] *= alpha;
      int row = g + 8 * (lane >> 4);
      Plds[w][row][lane & 15]        = f2bf(p0);
      Plds[w][row][16 + (lane & 15)] = f2bf(p1);
    }
    asm volatile("s_wait_dscnt 0" ::: "memory");

    int prow = lane & 15;
    int pc0 = (lane >> 4) << 3;
    FragA pf;
    pf.h[0] = *(const bf16x8*)&Plds[w][prow][pc0];
    pf.h[1] = *(const bf16x8*)&Plds[w][prow][pc0 + 16];
#pragma unroll
    for (int c = 0; c < 4; c++)
      O[c] = wmma_bf16(pf.v, load_frag(Vs, 32, c * 16, 0, lane), O[c]);

    __syncthreads();                  // all reads done before next tile overwrite
  }

  int lrow = 8 * (lane >> 4), lcol = lane & 15;
#pragma unroll
  for (int c = 0; c < 4; c++)
#pragma unroll
    for (int g = 0; g < 8; g++) {
      int row = q0 + g + lrow;
      int col = h * HD + c * 16 + lcol;
      float v = O[c][g] / lrun[g];
      ctx[(size_t)(b * SEQ + row) * HID + col] = f2bf(v);
    }
}

// ---------------------------------------------------------------------------
extern "C" void kernel_launch(void* const* d_in, const int* in_sizes, int n_in,
                              void* d_out, int out_size, void* d_ws, size_t ws_size,
                              hipStream_t stream) {
  (void)in_sizes; (void)n_in; (void)out_size; (void)ws_size;
  const float* text  = (const float*)d_in[0];
  const float* image = (const float*)d_in[1];
  const float* Wq = (const float*)d_in[2];
  const float* bq = (const float*)d_in[3];
  const float* Wk = (const float*)d_in[4];
  const float* bk = (const float*)d_in[5];
  const float* Wv = (const float*)d_in[6];
  const float* bv = (const float*)d_in[7];
  const float* Wo = (const float*)d_in[8];
  const float* bo = (const float*)d_in[9];
  const float* rel = (const float*)d_in[10];

  char* ws = (char*)d_ws;
  float* relmean = (float*)(ws + OFF_REL);
  u16* xbf  = (u16*)(ws + OFF_X);
  u16* wbf  = (u16*)(ws + OFF_W);
  u16* Qbf  = (u16*)(ws + OFF_Q);
  u16* Kbf  = (u16*)(ws + OFF_K);
  u16* Vtbf = (u16*)(ws + OFF_VT);
  u16* Ctx  = (u16*)(ws + OFF_CTX);

  rga_relmean_kernel<<<3, 256, 0, stream>>>(rel, relmean);

  int nx4 = (MTOT * HID) / 4;
  rga_convert_x_kernel<<<(nx4 + 255) / 256, 256, 0, stream>>>(text, image, xbf);
  int nw4 = (4 * WELEM) / 4;
  rga_convert_w_kernel<<<(nw4 + 255) / 256, 256, 0, stream>>>(Wq, Wk, Wv, Wo, wbf);

  dim3 gg(MTOT / 128, HID / 64);       // 64 x 12
  rga_gemm_kernel<0><<<gg, 256, 0, stream>>>(xbf, wbf + 0 * WELEM, bq, Qbf);
  rga_gemm_kernel<0><<<gg, 256, 0, stream>>>(xbf, wbf + 1 * WELEM, bk, Kbf);
  rga_gemm_kernel<1><<<gg, 256, 0, stream>>>(xbf, wbf + 2 * WELEM, bv, Vtbf);

  rga_attn_kernel<<<dim3(SEQ / 128, BATCH * HEADS), 256, 0, stream>>>(
      Qbf, Kbf, Vtbf, relmean, Ctx);

  rga_gemm_kernel<2><<<gg, 256, 0, stream>>>(Ctx, wbf + 3 * WELEM, bo, d_out);
}
